// Block_68212670595960
// MI455X (gfx1250) — compile-verified, matrix operand order checked by reading, not run on hardware
//
#include <hip/hip_runtime.h>
#include <stdint.h>

// ---------------------------------------------------------------------------
// Types for CDNA5 WMMA (wave32, 16x16x32 bf16 -> f32)
// ---------------------------------------------------------------------------
typedef __attribute__((ext_vector_type(16))) __bf16 v16bf;
typedef __attribute__((ext_vector_type(8)))  __bf16 v8bf;
typedef __attribute__((ext_vector_type(8)))  float  v8f;

union Frag { v16bf v; v8bf h[2]; };

__device__ __forceinline__ uint16_t f2bf(float f) {
  union { float f; uint32_t u; } c; c.f = f;
  uint32_t u = c.u;
  u += 0x7fffu + ((u >> 16) & 1u);   // round-to-nearest-even
  return (uint16_t)(u >> 16);
}

__device__ __forceinline__ v8f wmma_bf16(const Frag &a, const Frag &b, v8f c) {
  return __builtin_amdgcn_wmma_f32_16x16x32_bf16(
      /*neg_a=*/false, a.v, /*neg_b=*/false, b.v,
      /*c_mod=*/(short)0, c, /*reuse_a=*/false, /*reuse_b=*/false);
}

// CDNA5 async global->LDS copy (16B per lane), tracked by ASYNCcnt.
// VDST = VGPR holding per-lane LDS byte address, VADDR = 64-bit global addr.
__device__ __forceinline__ void async_g2l_b128(const void* g, void* l) {
  uint32_t laddr = (uint32_t)(size_t)l;     // low 32 bits of generic = LDS off
  asm volatile("global_load_async_to_lds_b128 %0, %1, off"
               :: "v"(laddr), "v"(g) : "memory");
}

// Problem constants
#define BB   4
#define SS   2048
#define DD   768
#define FF   3072
#define HH   12
#define HDIM 64
#define BS   (BB * SS)   // 8192 rows

// ---------------------------------------------------------------------------
// Weight convert + transpose: fp32 W[K][N] -> bf16 Wt[N][K]
// ---------------------------------------------------------------------------
__global__ void wtrans_kernel(const float* __restrict__ W,
                              uint16_t* __restrict__ Wt,
                              int Kdim, int Ndim) {
  size_t i = (size_t)blockIdx.x * 256 + threadIdx.x;
  size_t total = (size_t)Kdim * Ndim;
  if (i < total) {
    int n = (int)(i % Ndim);
    int k = (int)(i / Ndim);
    Wt[(size_t)n * Kdim + k] = f2bf(W[i]);
  }
}

// ---------------------------------------------------------------------------
// LayerNorm: fp32 [rows][768] -> bf16 [rows][768]; biased var, eps inside sqrt
// ---------------------------------------------------------------------------
__global__ void __launch_bounds__(256)
ln_kernel(const float* __restrict__ x, const float* __restrict__ w,
          const float* __restrict__ b, uint16_t* __restrict__ out) {
  __shared__ float red[256];
  int row = blockIdx.x;
  int tid = threadIdx.x;
  const float* xr = x + (size_t)row * DD;
  float v0 = xr[tid], v1 = xr[tid + 256], v2 = xr[tid + 512];

  float p = v0 + v1 + v2;
  red[tid] = p; __syncthreads();
  for (int s = 128; s > 0; s >>= 1) {
    if (tid < s) red[tid] += red[tid + s];
    __syncthreads();
  }
  float mean = red[0] * (1.0f / DD);
  __syncthreads();

  float c0 = v0 - mean, c1 = v1 - mean, c2 = v2 - mean;
  red[tid] = c0 * c0 + c1 * c1 + c2 * c2; __syncthreads();
  for (int s = 128; s > 0; s >>= 1) {
    if (tid < s) red[tid] += red[tid + s];
    __syncthreads();
  }
  float rstd = rsqrtf(red[0] * (1.0f / DD) + 1e-5f);

  uint16_t* o = out + (size_t)row * DD;
  o[tid]       = f2bf(c0 * rstd * w[tid]       + b[tid]);
  o[tid + 256] = f2bf(c1 * rstd * w[tid + 256] + b[tid + 256]);
  o[tid + 512] = f2bf(c2 * rstd * w[tid + 512] + b[tid + 512]);
}

// ---------------------------------------------------------------------------
// bf16 GEMM: C[M,N] = A[M,K](bf16,row-major) * Bt[N,K](bf16,row-major)^T
// K-slabs staged with GLOBAL_LOAD_ASYNC_TO_LDS_B128 (ASYNCcnt) and
// double-buffered: while computing slab p, slab p^1 is in flight.
// epilogue: (+bias)*scale, optional GELU(erf), optional fp32 residual,
// output layout per OUT_MODE:
//   0: bf16 row-major [M][N]
//   1: bf16 heads     [B,H,S,hd]   (for Q,K)
//   2: bf16 heads^T   [B,H,hd,S]   (for V)
//   3: fp32 row-major [M][N]
// Tile 128x64, K-step 32; 8 waves, each wave computes a 32x32 accumulator.
// ---------------------------------------------------------------------------
template <int OUT_MODE, bool GELU_F, bool RESID_F>
__global__ void __launch_bounds__(256)
gemm_bf16(const uint16_t* __restrict__ A, const uint16_t* __restrict__ Bt,
          const float* __restrict__ bias, const float* __restrict__ resid,
          float* __restrict__ outF, uint16_t* __restrict__ outB,
          int M, int N, int K, float scale) {
  (void)M;
  __shared__ __align__(16) uint16_t As[2][128 * 40];
  __shared__ __align__(16) uint16_t Bs[2][64 * 40];

  const int tid = threadIdx.x;
  const int lane = tid & 31, wv = tid >> 5;
  const int hi = lane >> 4, l15 = lane & 15;
  const int wm = wv & 3, wn = wv >> 2;          // 4 x 2 wave grid
  const int m0 = blockIdx.y * 128, n0 = blockIdx.x * 64;

  v8f acc[2][2];
  for (int i = 0; i < 2; ++i)
    for (int j = 0; j < 2; ++j) acc[i][j] = (v8f){0,0,0,0,0,0,0,0};

  const int ra = tid >> 1, sa = (tid & 1) * 16;  // A stage: 2 thr/row, 16 elems
  const int rb = tid >> 2, sb = (tid & 3) * 8;   // B stage: 4 thr/row,  8 elems

  // issue 3 async b128 copies for K-slab kk into buffer p
  auto issue_stage = [&](int kk, int p) {
    const uint16_t* ga = A + (size_t)(m0 + ra) * K + kk + sa;
    uint16_t* la = &As[p][ra * 40 + sa];
    async_g2l_b128(ga,     la);
    async_g2l_b128(ga + 8, la + 8);
    const uint16_t* gb = Bt + (size_t)(n0 + rb) * K + kk + sb;
    async_g2l_b128(gb, &Bs[p][rb * 40 + sb]);
  };

  issue_stage(0, 0);
  int p = 0;
  for (int kk = 0; kk < K; kk += 32) {
    const bool more = (kk + 32) < K;             // block-uniform
    if (more) {
      issue_stage(kk + 32, p ^ 1);
      asm volatile("s_wait_asynccnt 0x3" ::: "memory");  // slab p done
    } else {
      asm volatile("s_wait_asynccnt 0x0" ::: "memory");
    }
    __syncthreads();                              // all waves' copies visible

    Frag af[2], bf[2];
    for (int i = 0; i < 2; ++i) {
      int r = wm * 32 + i * 16 + l15;
      af[i].h[0] = *(const v8bf*)(&As[p][r * 40 + hi * 8]);
      af[i].h[1] = *(const v8bf*)(&As[p][r * 40 + 16 + hi * 8]);
    }
    for (int j = 0; j < 2; ++j) {
      int r = wn * 32 + j * 16 + l15;
      bf[j].h[0] = *(const v8bf*)(&Bs[p][r * 40 + hi * 8]);
      bf[j].h[1] = *(const v8bf*)(&Bs[p][r * 40 + 16 + hi * 8]);
    }
    for (int i = 0; i < 2; ++i)
      for (int j = 0; j < 2; ++j)
        acc[i][j] = wmma_bf16(af[i], bf[j], acc[i][j]);
    __syncthreads();                              // buffer p free for reuse
    p ^= 1;
  }

  // epilogue (C layout: lane holds col l15, rows e + 8*hi within 16x16 frag)
  for (int i = 0; i < 2; ++i) {
    for (int j = 0; j < 2; ++j) {
      int col = n0 + wn * 32 + j * 16 + l15;
      float bc = bias[col];
      for (int e = 0; e < 8; ++e) {
        int row = m0 + wm * 32 + i * 16 + e + 8 * hi;
        float v = (acc[i][j][e] + bc) * scale;
        if (GELU_F) v = 0.5f * v * (1.0f + erff(v * 0.70710678118654752f));
        if (RESID_F) v += resid[(size_t)row * N + col];
        if (OUT_MODE == 0) {
          outB[(size_t)row * N + col] = f2bf(v);
        } else if (OUT_MODE == 1) {
          int bidx = row >> 11, s = row & (SS - 1);
          int h = col >> 6, d = col & (HDIM - 1);
          outB[(((size_t)(bidx * HH + h)) * SS + s) * HDIM + d] = f2bf(v);
        } else if (OUT_MODE == 2) {
          int bidx = row >> 11, s = row & (SS - 1);
          int h = col >> 6, d = col & (HDIM - 1);
          outB[(((size_t)(bidx * HH + h)) * HDIM + d) * SS + s] = f2bf(v);
        } else {
          outF[(size_t)row * N + col] = v;
        }
      }
    }
  }
}

// ---------------------------------------------------------------------------
// Flash attention (causal), bf16 WMMA for QK^T and PV, fp32 online softmax.
// Q,K: [B,H,S,64] bf16 (Q pre-scaled by 1/sqrt(hd)); Vt: [B,H,64,S] bf16.
// One wave owns 16 query rows; 8 waves/block -> 128 queries per block.
// Key block = 32. O: bf16 [B*S, 768] row-major (merged heads).
// Q/K/Vt fragments load straight from global (layouts chosen so the
// B-fragment pattern == A-fragment pattern); only the P tile round-trips LDS.
// ---------------------------------------------------------------------------
__global__ void __launch_bounds__(256)
attn_kernel(const uint16_t* __restrict__ Q, const uint16_t* __restrict__ Km,
            const uint16_t* __restrict__ Vt, uint16_t* __restrict__ O) {
  __shared__ __align__(16) uint16_t Plds[8 * 16 * 40];

  const int b = blockIdx.z, h = blockIdx.y;
  const int tid = threadIdx.x;
  const int lane = tid & 31, wv = tid >> 5;
  const int hi = lane >> 4, l15 = lane & 15;
  const int qbase = blockIdx.x * 128 + wv * 16;

  const size_t headoff = ((size_t)(b * HH + h)) * SS * HDIM;
  const uint16_t* Qh  = Q  + headoff;
  const uint16_t* Kh  = Km + headoff;
  const uint16_t* Vth = Vt + headoff;            // [64][S]
  uint16_t* pl = Plds + wv * 16 * 40;            // per-wave P tile 16x32 (pad)

  // Q fragments (A layout), held in registers for the whole loop
  Frag qf[2];
  {
    const uint16_t* qr = Qh + (size_t)(qbase + l15) * HDIM;
    for (int f = 0; f < 2; ++f) {
      qf[f].h[0] = *(const v8bf*)(qr + f * 32 + hi * 8);
      qf[f].h[1] = *(const v8bf*)(qr + f * 32 + 16 + hi * 8);
    }
  }

  float mrow[8], lrow[8];
  v8f oacc[4];
  for (int e = 0; e < 8; ++e) { mrow[e] = -3.0e38f; lrow[e] = 0.0f; }
  for (int f = 0; f < 4; ++f) oacc[f] = (v8f){0,0,0,0,0,0,0,0};

  for (int kb = 0; kb < qbase + 16; kb += 32) {
    // ---- scores S = Q * K^T for 32 keys (2 N-frags, K-dim 64 = 2 steps)
    v8f s[2];
    s[0] = (v8f){0,0,0,0,0,0,0,0};
    s[1] = (v8f){0,0,0,0,0,0,0,0};
    for (int n = 0; n < 2; ++n) {
      const uint16_t* kr = Kh + (size_t)(kb + n * 16 + l15) * HDIM;
      for (int f = 0; f < 2; ++f) {
        Frag bk;
        bk.h[0] = *(const v8bf*)(kr + f * 32 + hi * 8);
        bk.h[1] = *(const v8bf*)(kr + f * 32 + 16 + hi * 8);
        s[n] = wmma_bf16(qf[f], bk, s[n]);
      }
    }
    // ---- causal mask
    if (kb + 31 > qbase) {
      for (int n = 0; n < 2; ++n) {
        int kc = kb + n * 16 + l15;
        for (int e = 0; e < 8; ++e) {
          int qr = qbase + e + 8 * hi;
          if (kc > qr) s[n][e] = -3.0e38f;
        }
      }
    }
    // ---- online softmax: row max / scale / exp / row sum (half-wave shfl)
    float al[8];
    for (int e = 0; e < 8; ++e) {
      float t = fmaxf(s[0][e], s[1][e]);
      t = fmaxf(t, __shfl_xor(t, 1, 32));
      t = fmaxf(t, __shfl_xor(t, 2, 32));
      t = fmaxf(t, __shfl_xor(t, 4, 32));
      t = fmaxf(t, __shfl_xor(t, 8, 32));
      float mnew = fmaxf(mrow[e], t);
      al[e] = __expf(mrow[e] - mnew);
      mrow[e] = mnew;
      s[0][e] = __expf(s[0][e] - mnew);
      s[1][e] = __expf(s[1][e] - mnew);
      float rs = s[0][e] + s[1][e];
      rs += __shfl_xor(rs, 1, 32);
      rs += __shfl_xor(rs, 2, 32);
      rs += __shfl_xor(rs, 4, 32);
      rs += __shfl_xor(rs, 8, 32);
      lrow[e] = lrow[e] * al[e] + rs;
    }
    for (int f = 0; f < 4; ++f)
      for (int e = 0; e < 8; ++e) oacc[f][e] *= al[e];

    // ---- P (C layout) -> LDS -> A-layout fragment
    for (int n = 0; n < 2; ++n)
      for (int e = 0; e < 8; ++e)
        pl[(e + 8 * hi) * 40 + n * 16 + l15] = f2bf(s[n][e]);
    asm volatile("s_wait_dscnt 0" ::: "memory");
    Frag ap;
    ap.h[0] = *(const v8bf*)(pl + l15 * 40 + hi * 8);
    ap.h[1] = *(const v8bf*)(pl + l15 * 40 + 16 + hi * 8);

    // ---- O += P * V   (B-frags straight from Vt[64][S])
    for (int f = 0; f < 4; ++f) {
      const uint16_t* vr = Vth + (size_t)(f * 16 + l15) * SS + kb;
      Frag bv;
      bv.h[0] = *(const v8bf*)(vr + hi * 8);
      bv.h[1] = *(const v8bf*)(vr + 16 + hi * 8);
      oacc[f] = wmma_bf16(ap, bv, oacc[f]);
    }
  }

  // ---- finalize and write merged-head bf16 [B*S][768]
  for (int e = 0; e < 8; ++e) {
    float inv = 1.0f / lrow[e];
    int grow = b * SS + qbase + e + 8 * hi;
    for (int f = 0; f < 4; ++f) {
      int col = h * HDIM + f * 16 + l15;
      O[(size_t)grow * DD + col] = f2bf(oacc[f][e] * inv);
    }
  }
}

// ---------------------------------------------------------------------------
// Host orchestration
// ---------------------------------------------------------------------------
extern "C" void kernel_launch(void* const* d_in, const int* in_sizes, int n_in,
                              void* d_out, int out_size, void* d_ws, size_t ws_size,
                              hipStream_t stream) {
  (void)in_sizes; (void)n_in; (void)out_size; (void)ws_size;
  const float* x     = (const float*)d_in[0];
  const float* Wq    = (const float*)d_in[1];
  const float* bq    = (const float*)d_in[2];
  const float* Wk    = (const float*)d_in[3];
  const float* bk    = (const float*)d_in[4];
  const float* Wv    = (const float*)d_in[5];
  const float* bv    = (const float*)d_in[6];
  const float* Wo    = (const float*)d_in[7];
  const float* bo    = (const float*)d_in[8];
  const float* ln1w  = (const float*)d_in[9];
  const float* ln1b  = (const float*)d_in[10];
  const float* ln2w  = (const float*)d_in[11];
  const float* ln2b  = (const float*)d_in[12];
  const float* W1    = (const float*)d_in[13];
  const float* b1    = (const float*)d_in[14];
  const float* W2    = (const float*)d_in[15];
  const float* b2    = (const float*)d_in[16];
  float* out = (float*)d_out;

  // workspace layout
  uint8_t* ws = (uint8_t*)d_ws;
  size_t off = 0;
  auto alloc = [&](size_t bytes) -> uint8_t* {
    uint8_t* p = ws + off;
    off = (off + bytes + 255) & ~(size_t)255;
    return p;
  };
  const size_t seg = (size_t)BS * DD;               // 6.29M elems
  uint16_t* wq_t = (uint16_t*)alloc((size_t)DD * DD * 2);
  uint16_t* wk_t = (uint16_t*)alloc((size_t)DD * DD * 2);
  uint16_t* wv_t = (uint16_t*)alloc((size_t)DD * DD * 2);
  uint16_t* wo_t = (uint16_t*)alloc((size_t)DD * DD * 2);
  uint16_t* w1_t = (uint16_t*)alloc((size_t)DD * FF * 2);
  uint16_t* w2_t = (uint16_t*)alloc((size_t)FF * DD * 2);
  uint16_t* hb   = (uint16_t*)alloc(seg * 2);       // ln1 out; reused for ln2
  float*    x1f  = (float*)   alloc(seg * 4);       // x + attn_out
  uint16_t* big  = (uint16_t*)alloc((size_t)BS * FF * 2); // qb|kb|vt|attn -> mb
  uint16_t* qb   = big;
  uint16_t* kb_  = big + seg;
  uint16_t* vt   = big + 2 * seg;
  uint16_t* attn = big + 3 * seg;
  uint16_t* mb   = big;                             // reuse after x1 computed

  // 1) weights -> bf16 transposed [N][K]
  {
    int tD = DD * DD, tF = DD * FF;
    wtrans_kernel<<<(tD + 255) / 256, 256, 0, stream>>>(Wq, wq_t, DD, DD);
    wtrans_kernel<<<(tD + 255) / 256, 256, 0, stream>>>(Wk, wk_t, DD, DD);
    wtrans_kernel<<<(tD + 255) / 256, 256, 0, stream>>>(Wv, wv_t, DD, DD);
    wtrans_kernel<<<(tD + 255) / 256, 256, 0, stream>>>(Wo, wo_t, DD, DD);
    wtrans_kernel<<<(tF + 255) / 256, 256, 0, stream>>>(W1, w1_t, DD, FF);
    wtrans_kernel<<<(tF + 255) / 256, 256, 0, stream>>>(W2, w2_t, FF, DD);
  }

  // 2) h = LN1(x) -> bf16
  ln_kernel<<<BS, 256, 0, stream>>>(x, ln1w, ln1b, hb);

  // 3) QKV projections (Q pre-scaled by 1/sqrt(64)=0.125)
  dim3 gD(DD / 64, BS / 128);
  gemm_bf16<1, false, false><<<gD, 256, 0, stream>>>(
      hb, wq_t, bq, nullptr, nullptr, qb, BS, DD, DD, 0.125f);
  gemm_bf16<1, false, false><<<gD, 256, 0, stream>>>(
      hb, wk_t, bk, nullptr, nullptr, kb_, BS, DD, DD, 1.0f);
  gemm_bf16<2, false, false><<<gD, 256, 0, stream>>>(
      hb, wv_t, bv, nullptr, nullptr, vt, BS, DD, DD, 1.0f);

  // 4) causal flash attention -> merged bf16 [BS][D]
  attn_kernel<<<dim3(SS / 128, HH, BB), 256, 0, stream>>>(qb, kb_, vt, attn);

  // 5) x1 = x + attn @ Wo + bo   (fp32)
  gemm_bf16<3, false, true><<<gD, 256, 0, stream>>>(
      attn, wo_t, bo, x, x1f, nullptr, BS, DD, DD, 1.0f);

  // 6) h2 = LN2(x1) -> bf16
  ln_kernel<<<BS, 256, 0, stream>>>(x1f, ln2w, ln2b, hb);

  // 7) m = GELU(h2 @ W1 + b1) -> bf16 [BS][F]
  dim3 gF(FF / 64, BS / 128);
  gemm_bf16<0, true, false><<<gF, 256, 0, stream>>>(
      hb, w1_t, b1, nullptr, nullptr, mb, BS, FF, DD, 1.0f);

  // 8) out = x1 + m @ W2 + b2   (fp32)
  gemm_bf16<3, false, true><<<gD, 256, 0, stream>>>(
      mb, w2_t, b2, x1f, out, nullptr, BS, DD, FF, 1.0f);
}
